// MultiTaskNet_46145128628683
// MI455X (gfx1250) — matrix operand order repaired and checked by reading, not compiled, not run on hardware
//
#include <hip/hip_runtime.h>
#include <hip/hip_bf16.h>

typedef __attribute__((ext_vector_type(16))) _Float16 v16h;
typedef __attribute__((ext_vector_type(8)))  float    v8f;

#define EMB 32

// x = [u | q | u*q]  (B x 96),  h = relu(x @ W1^T + b1)  (B x 64),
// score = relu(h @ W2^T + b2),  predictions = sum(u*q) + a + b.
// Wave-per-16-rows; layer 1 via v_wmma_f32_16x16x32_f16 (3 K-tiles x 4 N-tiles).
__global__ __launch_bounds__(256)
void mtnet_fused_kernel(const int*   __restrict__ user_ids,
                        const int*   __restrict__ item_ids,
                        const float* __restrict__ U_w,
                        const float* __restrict__ Q_w,
                        const float* __restrict__ A_w,
                        const float* __restrict__ B_w,
                        const float* __restrict__ W1,   // [64][96]
                        const float* __restrict__ b1,   // [64]
                        const float* __restrict__ W2,   // [1][64]
                        const float* __restrict__ b2,   // [1]
                        float* __restrict__ out_pred,   // [B]
                        float* __restrict__ out_score,  // [B]
                        int batch)
{
    const int lane = threadIdx.x & 31;
    const int wave = threadIdx.x >> 5;
    const int m16  = lane & 15;     // A: row within tile; B/C: column
    const int half = lane >> 4;     // K-half selector

    // ---- B fragments for W1^T (wave-invariant; built once, reused per tile) ----
    // B[k][n] = W1[n][k]; lane holds n = m16, frag[e] = W1[n][c*32 + half*8 + e + (e>=8 ? 8 : 0)]
    v16h bfrag[4][3];
    #pragma unroll
    for (int t = 0; t < 4; ++t) {
        const float* w1row = W1 + (t * 16 + m16) * 96;
        #pragma unroll
        for (int c = 0; c < 3; ++c) {
            const float* src = w1row + c * 32 + half * 8;
            #pragma unroll
            for (int e = 0; e < 16; ++e)
                bfrag[t][c][e] = (_Float16)src[e + (e >= 8 ? 8 : 0)];
        }
    }

    // layer-2 weights/biases for this lane's columns (n = t*16 + m16)
    float w2v[4], b1v[4];
    #pragma unroll
    for (int t = 0; t < 4; ++t) {
        w2v[t] = W2[t * 16 + m16];
        b1v[t] = b1[t * 16 + m16];
    }
    const float b2v = b2[0];

    const int tilesTotal = batch >> 4;
    const int wavesTotal = gridDim.x * (blockDim.x >> 5);
    const int waveId     = blockIdx.x * (blockDim.x >> 5) + wave;

    for (int tile = waveId; tile < tilesTotal; tile += wavesTotal) {
        const int rowBase = tile << 4;
        const int row = rowBase + m16;          // lanes m and m+16 share a row
        const int uid = user_ids[row];
        const int iid = item_ids[row];
        const float* urow = U_w + uid * EMB;
        const float* qrow = Q_w + iid * EMB;

        // A layout (f16 16x32): lane(m,half) holds K = half*8+0..7 and 16+half*8+0..7
        float uu[16], qq[16];
        {
            const float* ua = urow + half * 8;
            const float* ub = urow + 16 + half * 8;
            const float* qa = qrow + half * 8;
            const float* qb = qrow + 16 + half * 8;
            #pragma unroll
            for (int j = 0; j < 8; ++j) {
                uu[j] = ua[j]; uu[8 + j] = ub[j];
                qq[j] = qa[j]; qq[8 + j] = qb[j];
            }
        }

        v16h afrag[3];
        float dotp = 0.f;
        #pragma unroll
        for (int e = 0; e < 16; ++e) {
            float p = uu[e] * qq[e];
            dotp += p;
            afrag[0][e] = (_Float16)uu[e];   // K-tile 0: u
            afrag[1][e] = (_Float16)qq[e];   // K-tile 1: q
            afrag[2][e] = (_Float16)p;       // K-tile 2: u*q
        }

        // predictions: each half holds 16 of the 32 products for its row
        dotp += __shfl_xor(dotp, 16, 32);
        if (half == 0)
            out_pred[row] = dotp + A_w[uid] + B_w[iid];

        // ---- layer 1: 16x64 tile via 12 WMMAs ----
        v8f acc[4];
        #pragma unroll
        for (int t = 0; t < 4; ++t) {
            v8f c = {};
            #pragma unroll
            for (int kc = 0; kc < 3; ++kc)
                c = __builtin_amdgcn_wmma_f32_16x16x32_f16(
                        false, afrag[kc], false, bfrag[t][kc],
                        (short)0, c, false, false);
            acc[t] = c;
        }

        // ---- layer 2 on VALU: C/D layout row = half*8 + r, col = m16 ----
        // per-lane partials over the 4 column groups
        float p[8];
        #pragma unroll
        for (int r = 0; r < 8; ++r) {
            float s = 0.f;
            #pragma unroll
            for (int t = 0; t < 4; ++t) {
                float h = acc[t][r] + b1v[t];
                h = h > 0.f ? h : 0.f;       // relu
                s += h * w2v[t];
            }
            p[r] = s;
        }

        // Value-splitting butterfly: reduce 8 values across 16 columns with
        // 4+2+1+1 = 8 shuffles (vs 32 for the naive per-row butterfly).
        // After masks 8,4,2: lane m16 holds r = bit1(m16) + 2*bit2 + 4*bit3
        // summed over the 8 lanes sharing bit0; mask 1 completes the sum.
        #pragma unroll
        for (int j = 0; j < 4; ++j) {        // mask 8: 8 -> 4 values
            float lo = p[j], hi = p[j + 4];
            float send = (m16 & 8) ? lo : hi;
            float recv = __shfl_xor(send, 8, 32);
            p[j] = ((m16 & 8) ? hi : lo) + recv;
        }
        #pragma unroll
        for (int j = 0; j < 2; ++j) {        // mask 4: 4 -> 2 values
            float lo = p[j], hi = p[j + 2];
            float send = (m16 & 4) ? lo : hi;
            float recv = __shfl_xor(send, 4, 32);
            p[j] = ((m16 & 4) ? hi : lo) + recv;
        }
        {                                    // mask 2: 2 -> 1 value
            float lo = p[0], hi = p[1];
            float send = (m16 & 2) ? lo : hi;
            float recv = __shfl_xor(send, 2, 32);
            p[0] = ((m16 & 2) ? hi : lo) + recv;
        }
        p[0] += __shfl_xor(p[0], 1, 32);     // mask 1: finish 16-lane sum

        // even m16 lanes hold row r = m16>>1 (within this half)
        if ((m16 & 1) == 0) {
            float sc = p[0] + b2v;
            out_score[rowBase + half * 8 + (m16 >> 1)] = sc > 0.f ? sc : 0.f;
        }
    }
}

extern "C" void kernel_launch(void* const* d_in, const int* in_sizes, int n_in,
                              void* d_out, int out_size, void* d_ws, size_t ws_size,
                              hipStream_t stream) {
    const int*   user_ids = (const int*)  d_in[0];
    const int*   item_ids = (const int*)  d_in[1];
    const float* U_w      = (const float*)d_in[2];
    const float* Q_w      = (const float*)d_in[3];
    const float* A_w      = (const float*)d_in[4];
    const float* B_w      = (const float*)d_in[5];
    const float* W1       = (const float*)d_in[6];
    const float* b1       = (const float*)d_in[7];
    const float* W2       = (const float*)d_in[8];
    const float* b2       = (const float*)d_in[9];

    const int batch = in_sizes[0];              // 131072
    float* out_pred  = (float*)d_out;           // outputs concatenated: pred, score
    float* out_score = out_pred + batch;

    // 512 blocks x 8 waves = 4096 waves; 8192 tiles -> 2 tiles/wave
    // (amortizes B-fragment build; gather traffic ~35 MB is the bound)
    mtnet_fused_kernel<<<512, 256, 0, stream>>>(
        user_ids, item_ids, U_w, Q_w, A_w, B_w, W1, b1, W2, b2,
        out_pred, out_score, batch);
}
